// MyLoss_20959440405167
// MI455X (gfx1250) — compile-verified, matrix operand order checked by reading, not run on hardware
//
#include <hip/hip_runtime.h>
#include <hip/hip_fp16.h>

// ---------------- problem dims (fixed by the reference) ----------------
constexpr int kB   = 8;
constexpr int kC   = 64;
constexpr int kH   = 512;
constexpr int kW   = 512;
constexpr int kHW  = kH * kW;            // 262144 pixels per image
constexpr int TPB  = 256;                // 8 wave32 waves per block
constexpr int PPT  = 4;                  // pixels per thread (16B vector ops)
constexpr int TILE = TPB * PPT;          // 1024 pixels per block
constexpr int NBLK = kB * kHW / TILE;    // 2048 blocks (exact, no tail -> EXEC all ones)

typedef __attribute__((ext_vector_type(8))) int   v8i;
typedef __attribute__((ext_vector_type(4))) float f32x4;  // raw vector (not HIP_vector_type)
typedef int b128_t __attribute__((vector_size(16)));      // matches async builtin param type

// ---- gfx1250 async global->LDS path (ASYNCcnt), with safe fallback ----
#if __has_builtin(__builtin_amdgcn_global_load_async_to_lds_b128)
#define HAVE_ASYNC_LDS 1
// generic->AS1 is identity; generic->AS3 is low-32-bit truncation (matches the
// LLVM addrspacecast lowering and the flat-aperture layout), so int round-trips
// are a compile-safe way to produce the AS-qualified pointers the builtin wants.
#define ASYNC_COPY_B128(gsrc, ldst)                                                          \
  __builtin_amdgcn_global_load_async_to_lds_b128(                                            \
      (__attribute__((address_space(1))) b128_t*)(unsigned long long)(const void*)(gsrc),    \
      (__attribute__((address_space(3))) b128_t*)(unsigned)(unsigned long long)(const void*)(ldst), \
      0, 0)
#endif

#if __has_builtin(__builtin_amdgcn_s_wait_asynccnt)
#define WAIT_ASYNC(n) __builtin_amdgcn_s_wait_asynccnt(n)
#else
#define WAIT_ASYNC(n) asm volatile("s_wait_asynccnt %0" ::"n"(n) : "memory")
#endif

#if __has_builtin(__builtin_amdgcn_s_wait_dscnt)
#define WAIT_DS(n) __builtin_amdgcn_s_wait_dscnt(n)
#else
#define WAIT_DS(n) asm volatile("s_wait_dscnt %0" ::"n"(n) : "memory")
#endif

// counters: [0] = sum(|argmax-label| on valid), [1] = count(valid) over all (b,h,w)
__global__ __launch_bounds__(64) void init_counters_kernel(unsigned long long* __restrict__ cnt) {
  if (threadIdx.x < 2) cnt[threadIdx.x] = 0ull;
}

__global__ __launch_bounds__(TPB) void fused_argmax_l1_kernel(
    const float* __restrict__ outp,
    const int*  __restrict__ labels,
    const int*  __restrict__ mask,
    unsigned long long* __restrict__ cnt) {
  const int       t      = threadIdx.x;
  const long long gBase  = (long long)blockIdx.x * TILE;  // first flat (b,h,w) pixel of block
  const int       b      = (int)(gBase / kHW);
  const int       tileP0 = (int)(gBase % kHW);            // block never straddles images
  const int       p0     = tileP0 + PPT * t;              // this thread's pixel within image
  const size_t    src0   = (size_t)b * kC * kHW + (size_t)p0;  // + c*kHW per channel plane

  float bv0 = -INFINITY, bv1 = -INFINITY, bv2 = -INFINITY, bv3 = -INFINITY;
  int   bi0 = 0, bi1 = 0, bi2 = 0, bi3 = 0;

#if defined(HAVE_ASYNC_LDS)
  // Double-buffered async staging: fetch channel c+1 while scanning channel c.
  __shared__ __align__(16) float lds[2][TILE];
  ASYNC_COPY_B128(outp + src0, &lds[0][PPT * t]);  // prologue: channel 0
  int buf = 0;
  for (int c = 0; c < kC; ++c) {
    if (c + 1 < kC) {              // uniform branch -> EXEC stays all-ones
      WAIT_DS(0);                  // prior DS reads of lds[buf^1] retired before overwrite
      ASYNC_COPY_B128(outp + src0 + (size_t)(c + 1) * kHW, &lds[buf ^ 1][PPT * t]);
      WAIT_ASYNC(1);               // only the just-issued copy outstanding -> channel c landed
    } else {
      WAIT_ASYNC(0);
    }
    const f32x4 v = *(const f32x4*)&lds[buf][PPT * t];
    if (v.x > bv0) { bv0 = v.x; bi0 = c; }   // strict '>' keeps first max (jnp.argmax)
    if (v.y > bv1) { bv1 = v.y; bi1 = c; }
    if (v.z > bv2) { bv2 = v.z; bi2 = c; }
    if (v.w > bv3) { bv3 = v.w; bi3 = c; }
    buf ^= 1;
  }
#else
  for (int c = 0; c < kC; ++c) {
    const f32x4 v =
        __builtin_nontemporal_load((const f32x4*)(outp + src0 + (size_t)c * kHW));
    if (v.x > bv0) { bv0 = v.x; bi0 = c; }
    if (v.y > bv1) { bv1 = v.y; bi1 = c; }
    if (v.z > bv2) { bv2 = v.z; bi2 = c; }
    if (v.w > bv3) { bv3 = v.w; bi3 = c; }
  }
#endif

  const int4 lab = *(const int4*)(labels + gBase + PPT * t);
  const int4 m   = *(const int4*)(mask + p0);

  int d0 = bi0 - lab.x; d0 = d0 < 0 ? -d0 : d0;   // each <= 63, fits u8
  int d1 = bi1 - lab.y; d1 = d1 < 0 ? -d1 : d1;
  int d2 = bi2 - lab.z; d2 = d2 < 0 ? -d2 : d2;
  int d3 = bi3 - lab.w; d3 = d3 < 0 ? -d3 : d3;

  const unsigned packedD = (m.x ? (unsigned)d0 : 0u)
                         | ((m.y ? (unsigned)d1 : 0u) << 8)
                         | ((m.z ? (unsigned)d2 : 0u) << 16)
                         | ((m.w ? (unsigned)d3 : 0u) << 24);
  const unsigned packedV = (m.x ? 1u : 0u)
                         | ((m.y ? 1u : 0u) << 8)
                         | ((m.z ? 1u : 0u) << 16)
                         | ((m.w ? 1u : 0u) << 24);

  // Wave-level exact integer reduction via the matrix unit:
  // A(16x64 u8) holds the 128 packed bytes of the wave, B = all-ones -> C rows = row sums.
  const int one4 = 0x01010101;
  const v8i bones = { one4, one4, one4, one4, one4, one4, one4, one4 };
  const v8i czero = { 0, 0, 0, 0, 0, 0, 0, 0 };
  const v8i aD    = { (int)packedD, 0, 0, 0, 0, 0, 0, 0 };
  const v8i aV    = { (int)packedV, 0, 0, 0, 0, 0, 0, 0 };

  v8i cD = __builtin_amdgcn_wmma_i32_16x16x64_iu8(false, aD, false, bones, czero, false, false);
  v8i cV = __builtin_amdgcn_wmma_i32_16x16x64_iu8(false, aV, false, bones, czero, false, false);

  int sD = cD[0] + cD[1] + cD[2] + cD[3] + cD[4] + cD[5] + cD[6] + cD[7];  // rows 0..7 / 8..15
  int sV = cV[0] + cV[1] + cV[2] + cV[3] + cV[4] + cV[5] + cV[6] + cV[7];
  sD += __shfl_xor(sD, 16, 32);   // combine the two lane halves -> full wave sum
  sV += __shfl_xor(sV, 16, 32);

  if ((t & 31) == 0) {            // one atomic per wave
    atomicAdd(&cnt[0], (unsigned long long)(unsigned)sD);
    atomicAdd(&cnt[1], (unsigned long long)(unsigned)sV);
  }
}

// Reproduce the reference's fp16 semantics: int sums cast to f16 (overflow -> inf),
// divided in f16, result stored (as float in d_out).
__global__ __launch_bounds__(64) void finalize_kernel(
    const unsigned long long* __restrict__ cnt, float* __restrict__ out) {
  if (threadIdx.x == 0) {
    const float fs = __half2float(__float2half((float)cnt[0]));  // loss_sum as f16
    const float ft = __half2float(__float2half((float)cnt[1]));  // total as f16
    out[0] = __half2float(__float2half(fs / ft));
  }
}

extern "C" void kernel_launch(void* const* d_in, const int* in_sizes, int n_in,
                              void* d_out, int out_size, void* d_ws, size_t ws_size,
                              hipStream_t stream) {
  (void)in_sizes; (void)n_in; (void)out_size; (void)ws_size;
  const float* outputs = (const float*)d_in[0];
  const int*   labels  = (const int*)d_in[1];
  const int*   maskp   = (const int*)d_in[2];
  unsigned long long* cnt = (unsigned long long*)d_ws;
  float* out = (float*)d_out;

  init_counters_kernel<<<dim3(1), dim3(64), 0, stream>>>(cnt);
  fused_argmax_l1_kernel<<<dim3(NBLK), dim3(TPB), 0, stream>>>(outputs, labels, maskp, cnt);
  finalize_kernel<<<dim3(1), dim3(64), 0, stream>>>(cnt, out);
}
